// TaylorKANLinear_17411797418706
// MI455X (gfx1250) — matrix operand order repaired and checked by reading, not compiled, not run on hardware
//
#include <hip/hip_runtime.h>
#include <stdint.h>

// ---------------------------------------------------------------------------
// TaylorKAN linear == one bf16 WMMA GEMM after algebraic restructuring:
//   out = F[8192 x 4096] @ Wb[1024 x 4096]^T + bias'
//   F[b,4i+k]  = {silu(x), x, x^2, x^3}   (bf16)
//   Wb[o,4i+k] = {W[o,i], C[o,i,1..3]}    (bf16)
//   bias'[o]   = bias[o] + sum_i C[o,i,0]     (x^0 term folded)
// GEMM: 128x128 block tile, 4 waves, 64x64 wave tile (16 WMMA / K-step),
// double-buffered LDS fed by GLOBAL_LOAD_ASYNC_TO_LDS_B128 (ASYNCcnt DMA).
// ---------------------------------------------------------------------------

#define B_DIM   8192
#define IN_DIM  1024
#define OUT_DIM 1024
#define K_DIM   (IN_DIM * 4)   // 4096

#define BM 128
#define BN 128
#define BK 32
#define LDS_LD 40              // 32 + 8 pad (bf16 elems); 80B row stride, 16B aligned

typedef __bf16 v16bf __attribute__((ext_vector_type(16)));
typedef float  v8f   __attribute__((ext_vector_type(8)));
typedef int    v4i   __attribute__((ext_vector_type(4)));

#if __has_builtin(__builtin_amdgcn_global_load_async_to_lds_b128)
#define HAVE_ASYNC_LDS 1
typedef __attribute__((address_space(1))) v4i glb_v4i;   // AS1 ("__device__") int4
typedef __attribute__((address_space(3))) v4i lds_v4i;   // AS3 ("__shared__") int4
#else
#define HAVE_ASYNC_LDS 0
#endif

__device__ __forceinline__ unsigned short f2bf(float f) {
    union { float f; unsigned u; } c; c.f = f;
    unsigned u = c.u;
    unsigned r = u + 0x7FFFu + ((u >> 16) & 1u);   // round-to-nearest-even
    return (unsigned short)(r >> 16);
}

// Stage one 64B tile row (32 bf16) from global into LDS.
__device__ __forceinline__ void stage64(const unsigned short* __restrict__ g,
                                        unsigned short* l) {
#if HAVE_ASYNC_LDS
#pragma unroll
    for (int j = 0; j < 4; ++j)
        __builtin_amdgcn_global_load_async_to_lds_b128(
            (glb_v4i*)(g + j * 8), (lds_v4i*)(l + j * 8), 0, 0);
#else
    uint4 r0 = ((const uint4*)g)[0];
    uint4 r1 = ((const uint4*)g)[1];
    uint4 r2 = ((const uint4*)g)[2];
    uint4 r3 = ((const uint4*)g)[3];
    ((uint4*)l)[0] = r0; ((uint4*)l)[1] = r1;
    ((uint4*)l)[2] = r2; ((uint4*)l)[3] = r3;
#endif
}

__device__ __forceinline__ void wait_stage() {
#if HAVE_ASYNC_LDS
#if __has_builtin(__builtin_amdgcn_s_wait_asynccnt)
    __builtin_amdgcn_s_wait_asynccnt(0);
#else
    asm volatile("s_wait_asynccnt 0x0" ::: "memory");
#endif
#endif
}

// ---- Kernel 1: F[b, 4i+k] = {silu(x), x, x^2, x^3} in bf16
__global__ __launch_bounds__(256)
void feat_kernel(const float* __restrict__ x, unsigned short* __restrict__ F) {
    int idx = blockIdx.x * 256 + threadIdx.x;       // one (b,i) pair per thread
    float v  = x[idx];
    float s  = v / (1.0f + __expf(-v));             // SiLU
    float v2 = v * v;
    float v3 = v2 * v;
    ushort4 o;
    o.x = f2bf(s); o.y = f2bf(v); o.z = f2bf(v2); o.w = f2bf(v3);
    *(ushort4*)(F + (size_t)idx * 4) = o;           // coalesced 8B stores
}

// ---- Kernel 2: Wb[o, 4i+k] = {W[o,i], C[o,i,1], C[o,i,2], C[o,i,3]} in bf16
__global__ __launch_bounds__(256)
void pack_kernel(const float* __restrict__ W, const float* __restrict__ C,
                 unsigned short* __restrict__ Wb) {
    int idx = blockIdx.x * 256 + threadIdx.x;       // one (o,i) pair per thread
    const float* c = C + (size_t)idx * 4;
    ushort4 o;
    o.x = f2bf(W[idx]);
    o.y = f2bf(c[1]); o.z = f2bf(c[2]); o.w = f2bf(c[3]);
    *(ushort4*)(Wb + (size_t)idx * 4) = o;
}

// ---- Kernel 3: bias'[o] = bias[o] + sum_i C[o,i,0]
__global__ __launch_bounds__(256)
void bias_kernel(const float* __restrict__ C, const float* __restrict__ bias,
                 float* __restrict__ biasAdj) {
    int o = blockIdx.x;
    __shared__ float red[256];
    float s = 0.f;
    for (int i = threadIdx.x; i < IN_DIM; i += 256)
        s += C[((size_t)o * IN_DIM + i) * 4 + 0];
    red[threadIdx.x] = s;
    __syncthreads();
    for (int st = 128; st > 0; st >>= 1) {
        if (threadIdx.x < st) red[threadIdx.x] += red[threadIdx.x + st];
        __syncthreads();
    }
    if (threadIdx.x == 0) biasAdj[o] = bias[o] + red[0];
}

// ---- Kernel 4: bf16 WMMA GEMM. 128 threads = 4 waves, wave grid 2x2,
// each wave computes a 64x64 output patch = 4x4 tiles of 16x16.
__global__ __launch_bounds__(128)
void gemm_kernel(const unsigned short* __restrict__ F,
                 const unsigned short* __restrict__ Wb,
                 const float* __restrict__ biasAdj,
                 float* __restrict__ out) {
    __shared__ __align__(16) unsigned short sA[2][BM * LDS_LD];
    __shared__ __align__(16) unsigned short sB[2][BN * LDS_LD];

    const int tid  = threadIdx.x;                  // 0..127
    const int lane = tid & 31;
    const int wave = tid >> 5;                     // 0..3
    const int wm   = wave >> 1;                    // 0..1 : 64 rows of M
    const int wn   = wave & 1;                     // 0..1 : 64 cols of N
    const int half = lane >> 4;                    // high-K / high-M lane group
    const int l16  = lane & 15;

    const int row0 = blockIdx.y * BM;
    const int col0 = blockIdx.x * BN;

    // staging: one full tile row (32 bf16 = 64B) per thread, rows 0..127
    const unsigned short* gA = F  + (size_t)(row0 + tid) * K_DIM;
    const unsigned short* gB = Wb + (size_t)(col0 + tid) * K_DIM;
    const int stOff = tid * LDS_LD;

    // prologue: DMA tile 0 into buffer 0
    stage64(gA, &sA[0][stOff]);
    stage64(gB, &sB[0][stOff]);
    wait_stage();
    __syncthreads();

    v8f acc[4][4];
#pragma unroll
    for (int mi = 0; mi < 4; ++mi)
#pragma unroll
        for (int ni = 0; ni < 4; ++ni)
            acc[mi][ni] = (v8f){0.f, 0.f, 0.f, 0.f, 0.f, 0.f, 0.f, 0.f};

    union Frag { uint4 q[2]; v16bf v; };

    int buf = 0;
    for (int k0 = 0; k0 < K_DIM; k0 += BK) {
        const int nk   = k0 + BK;
        const bool more = (nk < K_DIM);
        if (more) {                                // DMA next tile under WMMA shadow
            const int nb = buf ^ 1;
            stage64(gA + nk, &sA[nb][stOff]);
            stage64(gB + nk, &sB[nb][stOff]);
        }

        // Fragment gathers per ISA 7.12.2 VGPR layouts.
        Frag a[4], b[4];
#pragma unroll
        for (int mi = 0; mi < 4; ++mi) {
            // A 16x32 bf16: lane row m=l16; VGPR0-3: K=8*half+0..7 ; VGPR4-7: K=16+8*half+0..7
            const unsigned short* p =
                &sA[buf][(wm * 64 + mi * 16 + l16) * LDS_LD + half * 8];
            a[mi].q[0] = *(const uint4*)(p);
            a[mi].q[1] = *(const uint4*)(p + 16);
        }
#pragma unroll
        for (int ni = 0; ni < 4; ++ni) {
            // B 32x16 bf16: lane col n=l16; VGPR0-3: K=8*half+0..7 ; VGPR4-7: K=16+8*half+0..7
            const unsigned short* p =
                &sB[buf][(wn * 64 + ni * 16 + l16) * LDS_LD + half * 8];
            b[ni].q[0] = *(const uint4*)(p);
            b[ni].q[1] = *(const uint4*)(p + 16);
        }

#pragma unroll
        for (int mi = 0; mi < 4; ++mi)
#pragma unroll
            for (int ni = 0; ni < 4; ++ni)      // consecutive WMMAs share A operand
                acc[mi][ni] = __builtin_amdgcn_wmma_f32_16x16x32_bf16(
                    false, a[mi].v, false, b[ni].v,
                    (short)0, acc[mi][ni], false, false);

        if (more) {
            wait_stage();                          // my DMA into buf^1 done
            __syncthreads();                       // everyone's DMA done + buf reads done
            buf ^= 1;
        }
    }

    // Epilogue: C/D layout -> row = 8*half + r, col = l16; one bias scalar per lane/N-tile
#pragma unroll
    for (int ni = 0; ni < 4; ++ni) {
        const int col = col0 + wn * 64 + ni * 16 + l16;
        const float bv = biasAdj[col];
#pragma unroll
        for (int mi = 0; mi < 4; ++mi) {
            const int rbase = row0 + wm * 64 + mi * 16 + half * 8;
            v8f c = acc[mi][ni];
#pragma unroll
            for (int r = 0; r < 8; ++r)
                out[(size_t)(rbase + r) * OUT_DIM + col] = c[r] + bv;
        }
    }
}

extern "C" void kernel_launch(void* const* d_in, const int* in_sizes, int n_in,
                              void* d_out, int out_size, void* d_ws, size_t ws_size,
                              hipStream_t stream) {
    const float* x    = (const float*)d_in[0];   // [8192,1024]
    const float* W    = (const float*)d_in[1];   // [1024,1024]
    const float* C    = (const float*)d_in[2];   // [1024,1024,4]
    const float* bias = (const float*)d_in[3];   // [1024]
    float* out = (float*)d_out;

    // workspace: F (64 MB bf16) | Wb (8 MB bf16) | bias' (4 KB f32)
    unsigned short* F  = (unsigned short*)d_ws;
    unsigned short* Wb = (unsigned short*)((char*)d_ws + (size_t)B_DIM * K_DIM * 2);
    float* biasAdj     = (float*)((char*)d_ws + (size_t)B_DIM * K_DIM * 2
                                              + (size_t)OUT_DIM * K_DIM * 2);

    feat_kernel<<<(B_DIM * IN_DIM) / 256, 256, 0, stream>>>(x, F);
    pack_kernel<<<(OUT_DIM * IN_DIM) / 256, 256, 0, stream>>>(W, C, Wb);
    bias_kernel<<<OUT_DIM, 256, 0, stream>>>(C, bias, biasAdj);

    dim3 grid(OUT_DIM / BN, B_DIM / BM);         // 8 x 64 blocks
    gemm_kernel<<<grid, 128, 0, stream>>>(F, Wb, biasAdj, out);
}
